// cgcnn_11458972746252
// MI455X (gfx1250) — matrix operand order repaired and checked by reading, not compiled, not run on hardware
//
#include <hip/hip_runtime.h>
#include <math.h>

// ---------------- problem constants ----------------
#define Nn   50000
#define Ee   800000
#define INF_ 92
#define EDF  50
#define Hf   64
#define Lf   4
#define Gf   512
#define EPSb 1e-5f

typedef unsigned int  uint32;
typedef unsigned short ushort16;

typedef __attribute__((ext_vector_type(16))) __bf16 v16bf;
typedef __attribute__((ext_vector_type(8)))  float  v8f;
typedef __attribute__((ext_vector_type(4)))  uint32 v4u;

union Frag16 {
    v16bf v;
    uint32 u[8];
    v4u    q[2];
};

__device__ __forceinline__ ushort16 f2bf(float f) {
    uint32 u = __float_as_uint(f);
    uint32 r = u + 0x7FFFu + ((u >> 16) & 1u);   // round-to-nearest-even
    return (ushort16)(r >> 16);
}

// ---------------- generic fill ----------------
__global__ void fill_k(float* p, float v, int n) {
    int i = blockIdx.x * blockDim.x + threadIdx.x;
    if (i < n) p[i] = v;
}

// ---------------- node embed: h = relu(x @ W + b), hb = bf16(h) -------------
__global__ void node_embed_k(const float* __restrict__ x,
                             const float* __restrict__ W,
                             const float* __restrict__ b,
                             float* __restrict__ h, ushort16* __restrict__ hb) {
    int idx = blockIdx.x * blockDim.x + threadIdx.x;
    if (idx >= Nn * Hf) return;
    int n = idx >> 6, c = idx & 63;
    float acc = b[c];
    const float* xr = x + (size_t)n * INF_;
    #pragma unroll 4
    for (int k = 0; k < INF_; ++k) acc += xr[k] * W[k * Hf + c];
    float v = fmaxf(acc, 0.f);
    h[idx] = v;
    hb[idx] = f2bf(v);
}

// -------- swizzle conv weights into WMMA-B fragment layout (bf16 pairs) -----
// dest index: ((((l*2+mat)*6 + c)*4 + t)*32 + lane)*8 + i
// value     : pack(bf16 W[k][col], bf16 W[k+1][col]), k = c*32 + (lane>>4)*16 + 2i
__global__ void wswz_k(const float* __restrict__ Wsm,
                       const float* __restrict__ Wcv,
                       uint32* __restrict__ wq) {
    int idx = blockIdx.x * blockDim.x + threadIdx.x;
    const int TOT = Lf * 2 * 6 * 4 * 32 * 8;
    if (idx >= TOT) return;
    int i    = idx & 7;
    int lane = (idx >> 3) & 31;
    int t    = (idx >> 8) & 3;
    int c    = (idx >> 10) & 7;       // 0..5
    int lm   = idx >> 13;             // l*2 + mat
    int mat  = lm & 1, l = lm >> 1;
    int k   = c * 32 + (lane >> 4) * 16 + 2 * i;
    int col = t * 16 + (lane & 15);
    const float* W = mat ? Wcv : Wsm;
    size_t base = ((size_t)l * 192 + k) * Hf + col;
    uint32 lo = f2bf(W[base]);
    uint32 hi = f2bf(W[base + Hf]);
    wq[idx] = lo | (hi << 16);
}

// -------- swizzle pre_edge_w [50,64] into B-fragment layout, K padded to 64 --
__global__ void pewswz_k(const float* __restrict__ W, uint32* __restrict__ wq) {
    int idx = blockIdx.x * blockDim.x + threadIdx.x;
    const int TOT = 2 * 4 * 32 * 8;   // 2 K-chunks
    if (idx >= TOT) return;
    int i    = idx & 7;
    int lane = (idx >> 3) & 31;
    int t    = (idx >> 8) & 3;
    int c    = idx >> 10;             // 0..1
    int k   = c * 32 + (lane >> 4) * 16 + 2 * i;
    int col = t * 16 + (lane & 15);
    uint32 lo = (k     < EDF) ? (uint32)f2bf(W[(size_t)k * Hf + col])       : 0u;
    uint32 hi = (k + 1 < EDF) ? (uint32)f2bf(W[(size_t)(k + 1) * Hf + col]) : 0u;
    wq[idx] = lo | (hi << 16);
}

// ---------------- edge embed (WMMA): eb = bf16(relu(ea @ W + b)) ------------
// block = 128 threads = 4 waves; each wave -> one 16-edge tile; K=64 (padded)
__global__ __launch_bounds__(128)
void edge_embed_wmma_k(const float* __restrict__ ea,
                       const uint32* __restrict__ wq,   // [2*4*32*8] swizzled
                       const float* __restrict__ b,
                       ushort16* __restrict__ eb) {
    __shared__ uint32   wpe[2 * 4 * 32 * 8];            // 8 KB
    __shared__ ushort16 ze[4][16 * 64];                 // 8 KB

    const int tid  = threadIdx.x;
    const int wave = tid >> 5;
    const int lane = tid & 31;

    for (int i = tid; i < 2 * 4 * 32 * 8; i += 128) wpe[i] = wq[i];

    const int e0 = blockIdx.x * 64 + wave * 16;
    for (int r = 0; r < 16; ++r) {
        const float* src = ea + (size_t)(e0 + r) * EDF;
        #pragma unroll
        for (int j = lane; j < 64; j += 32)
            ze[wave][r * 64 + j] = (j < EDF) ? f2bf(src[j]) : (ushort16)0;
    }
    __syncthreads();

    v8f acc[4] = {};
    const int half = lane >> 4;
    const int mrow = lane & 15;

    #pragma unroll
    for (int c = 0; c < 2; ++c) {
        Frag16 A;
        const ushort16* zr = &ze[wave][mrow * 64 + c * 32];
        #pragma unroll
        for (int i = 0; i < 8; ++i) {
            int kk = 2 * i + 16 * half;
            A.u[i] = *(const uint32*)(zr + kk);
        }
        #pragma unroll
        for (int t = 0; t < 4; ++t) {
            int base = ((c * 4 + t) * 32 + lane) * 8;
            Frag16 B;
            B.q[0] = *(const v4u*)&wpe[base];
            B.q[1] = *(const v4u*)&wpe[base + 4];
            acc[t] = __builtin_amdgcn_wmma_f32_16x16x32_bf16(
                         false, A.v, false, B.v, (short)0, acc[t], false, false);
        }
    }

    const int ncol = lane & 15;
    #pragma unroll
    for (int r = 0; r < 8; ++r) {
        int m = r + 8 * half;
        size_t rowb = (size_t)(e0 + m) * Hf;
        #pragma unroll
        for (int t = 0; t < 4; ++t) {
            int col = t * 16 + ncol;
            float v = fmaxf(acc[t][r] + b[col], 0.f);
            eb[rowb + col] = f2bf(v);
        }
    }
}

// ---------------- main WMMA edge-conv kernel ----------------
// block = 128 threads = 4 waves; each wave -> TPW sequential 16-edge tiles,
// amortizing the 48KB LDS weight stage over 256 edges per block.
#define CWAVES 4
#define TPW    4
__global__ __launch_bounds__(128)
void edge_conv_k(const int* __restrict__ ei,          // [2][E]: src row0, dst row1
                 const ushort16* __restrict__ hb,     // [N][64] bf16
                 const ushort16* __restrict__ eb,     // [E][64] bf16
                 const uint32* __restrict__ wq_l,     // layer weights: [2][6144] uints
                 const float* __restrict__ smb,       // [64]
                 const float* __restrict__ cvb,       // [64]
                 float* __restrict__ agg) {           // [N][64] fp32 (atomic)
    __shared__ uint32   wsm[6 * 4 * 32 * 8];          // 24576 B
    __shared__ uint32   wcv[6 * 4 * 32 * 8];          // 24576 B
    __shared__ ushort16 zt[CWAVES][16 * 192];         // 24576 B
    __shared__ int      dstb[CWAVES][16];

    const int tid  = threadIdx.x;
    const int wave = tid >> 5;
    const int lane = tid & 31;
    const int half = lane >> 4;
    const int mrow = lane & 15;
    const int ncol = lane & 15;

    // stage both weight matrices once per block
    for (int i = tid; i < 6144; i += 128) {
        wsm[i] = wq_l[i];
        wcv[i] = wq_l[6144 + i];
    }
    __syncthreads();

    // per-lane bias values (fixed columns for this lane)
    float bS[4], bC[4];
    #pragma unroll
    for (int t = 0; t < 4; ++t) {
        bS[t] = smb[t * 16 + ncol];
        bC[t] = cvb[t * 16 + ncol];
    }

    for (int it = 0; it < TPW; ++it) {
        const int e0 = (blockIdx.x * (CWAVES * TPW) + wave * TPW + it) * 16;

        // stage z tile (wave-local; LDS ops are in-order per wave)
        for (int r = 0; r < 16; ++r) {
            int e = e0 + r;
            int s = ei[e];
            int d = ei[Ee + e];
            if (lane == 0) dstb[wave][r] = d;
            const uint32* hd = (const uint32*)(hb + (size_t)d * Hf);
            const uint32* hs = (const uint32*)(hb + (size_t)s * Hf);
            const uint32* ee = (const uint32*)(eb + (size_t)e * Hf);
            uint32* zr = (uint32*)&zt[wave][r * 192];
            #pragma unroll
            for (int j = lane; j < 96; j += 32) {
                uint32 v;
                if (j < 32)      v = hd[j];
                else if (j < 64) v = hs[j - 32];
                else             v = ee[j - 64];
                zr[j] = v;
            }
        }

        v8f accS[4] = {};
        v8f accC[4] = {};

        #pragma unroll
        for (int c = 0; c < 6; ++c) {
            // A fragment: 16x32 bf16, contiguous k = 16*half + s (pairs contiguous in z row)
            Frag16 A;
            const ushort16* zr = &zt[wave][mrow * 192 + c * 32];
            #pragma unroll
            for (int i = 0; i < 8; ++i) {
                int kk = 2 * i + 16 * half;
                A.u[i] = *(const uint32*)(zr + kk);
            }
            #pragma unroll
            for (int t = 0; t < 4; ++t) {
                int base = ((c * 4 + t) * 32 + lane) * 8;
                Frag16 Bs, Bc;
                Bs.q[0] = *(const v4u*)&wsm[base];
                Bs.q[1] = *(const v4u*)&wsm[base + 4];
                Bc.q[0] = *(const v4u*)&wcv[base];
                Bc.q[1] = *(const v4u*)&wcv[base + 4];
                accS[t] = __builtin_amdgcn_wmma_f32_16x16x32_bf16(
                              false, A.v, false, Bs.v, (short)0, accS[t], false, false);
                accC[t] = __builtin_amdgcn_wmma_f32_16x16x32_bf16(
                              false, A.v, false, Bc.v, (short)0, accC[t], false, false);
            }
        }

        // epilogue: bias + softplus*sigmoid, scatter-add by dst
        #pragma unroll
        for (int r = 0; r < 8; ++r) {
            int m = r + 8 * half;
            int d = dstb[wave][m];
            float* arow = agg + (size_t)d * Hf;
            #pragma unroll
            for (int t = 0; t < 4; ++t) {
                float vs = accS[t][r] + bS[t];
                float vc = accC[t][r] + bC[t];
                float sp = fmaxf(vs, 0.f) + log1pf(expf(-fabsf(vs)));
                float sg = 1.f / (1.f + expf(-vc));
                atomicAdd(arow + (t * 16 + ncol), sp * sg);
            }
        }
    }
}

// ---------------- per-column sum / sumsq over agg ----------------
#define STAT_BLOCKS 200
__global__ __launch_bounds__(256)
void colstats_k(const float* __restrict__ agg,
                float* __restrict__ sums, float* __restrict__ sumsq) {
    __shared__ float rs[256], rq[256];
    int t   = threadIdx.x;
    int col = t & 63;
    int rg  = t >> 6;                    // 0..3
    int rows = Nn / STAT_BLOCKS;         // 250
    int r0 = blockIdx.x * rows;
    float s = 0.f, q = 0.f;
    for (int r = r0 + rg; r < r0 + rows; r += 4) {
        float v = agg[(size_t)r * Hf + col];
        s += v;
        q += v * v;
    }
    rs[t] = s; rq[t] = q;
    __syncthreads();
    if (rg == 0) {
        s = rs[col] + rs[col + 64] + rs[col + 128] + rs[col + 192];
        q = rq[col] + rq[col + 64] + rq[col + 128] + rq[col + 192];
        atomicAdd(&sums[col], s);
        atomicAdd(&sumsq[col], q);
    }
}

__global__ void finstats_k(const float* sums, const float* sumsq,
                           float* mu, float* rsig) {
    int c = threadIdx.x;
    if (c < 64) {
        float m = sums[c] / (float)Nn;
        float v = sumsq[c] / (float)Nn - m * m;
        mu[c]   = m;
        rsig[c] = rsqrtf(v + EPSb);
    }
}

// ---------------- BN + residual, refresh bf16 node features ----------------
__global__ void bnres_k(const float* __restrict__ agg,
                        const float* __restrict__ mu, const float* __restrict__ rsig,
                        const float* __restrict__ gamma, const float* __restrict__ beta,
                        float* __restrict__ h, ushort16* __restrict__ hb) {
    int idx = blockIdx.x * blockDim.x + threadIdx.x;
    if (idx >= Nn * Hf) return;
    int c = idx & 63;
    float v = (agg[idx] - mu[c]) * rsig[c] * gamma[c] + beta[c] + h[idx];
    h[idx]  = v;
    hb[idx] = f2bf(v);
}

// ---------------- pre-pool dense: pp = relu(h @ W + b) ----------------
__global__ void prepool_k(const float* __restrict__ h,
                          const float* __restrict__ W,
                          const float* __restrict__ b,
                          float* __restrict__ pp) {
    int idx = blockIdx.x * blockDim.x + threadIdx.x;
    if (idx >= Nn * Hf) return;
    int n = idx >> 6, c = idx & 63;
    float acc = b[c];
    const float* hr = h + (size_t)n * Hf;
    #pragma unroll 8
    for (int k = 0; k < Hf; ++k) acc += hr[k] * W[k * Hf + c];
    pp[idx] = fmaxf(acc, 0.f);
}

// ---------------- segment sum pool + counts ----------------
__global__ void pool_k(const float* __restrict__ pp, const int* __restrict__ batch,
                       float* __restrict__ pool, float* __restrict__ cnt) {
    int idx = blockIdx.x * blockDim.x + threadIdx.x;
    if (idx >= Nn * Hf) return;
    int n = idx >> 6, c = idx & 63;
    int g = batch[n];
    atomicAdd(&pool[(size_t)g * Hf + c], pp[idx]);
    if (c == 0) atomicAdd(&cnt[g], 1.f);
}

// ---------------- per-graph MLP head ----------------
__global__ __launch_bounds__(64)
void head_k(const float* __restrict__ pool, const float* __restrict__ cnt,
            const float* __restrict__ hw, const float* __restrict__ hbias,
            const float* __restrict__ ow, const float* __restrict__ ob,
            float* __restrict__ out) {
    __shared__ float xs[64];
    __shared__ float red[64];
    int g = blockIdx.x;
    int t = threadIdx.x;
    float c = fmaxf(cnt[g], 1.f);
    xs[t] = pool[(size_t)g * Hf + t] / c;
    __syncthreads();
    for (int j = 0; j < 2; ++j) {
        float acc = hbias[j * Hf + t];
        #pragma unroll 8
        for (int k = 0; k < Hf; ++k)
            acc += xs[k] * hw[((size_t)j * Hf + k) * Hf + t];
        float v = fmaxf(acc, 0.f);
        __syncthreads();
        xs[t] = v;
        __syncthreads();
    }
    red[t] = xs[t] * ow[t];
    __syncthreads();
    for (int s = 32; s > 0; s >>= 1) {
        if (t < s) red[t] += red[t + s];
        __syncthreads();
    }
    if (t == 0) out[g] = red[0] + ob[0];
}

// ---------------- launch ----------------
extern "C" void kernel_launch(void* const* d_in, const int* in_sizes, int n_in,
                              void* d_out, int out_size, void* d_ws, size_t ws_size,
                              hipStream_t stream) {
    const float* x      = (const float*)d_in[0];
    const int*   ei     = (const int*)d_in[1];
    const float* ea     = (const float*)d_in[2];
    const int*   batch  = (const int*)d_in[3];
    const float* pnw    = (const float*)d_in[4];
    const float* pnb    = (const float*)d_in[5];
    const float* pew    = (const float*)d_in[6];
    const float* peb    = (const float*)d_in[7];
    const float* csw    = (const float*)d_in[8];
    const float* csb    = (const float*)d_in[9];
    const float* ccw    = (const float*)d_in[10];
    const float* ccb    = (const float*)d_in[11];
    const float* gamma  = (const float*)d_in[12];
    const float* beta   = (const float*)d_in[13];
    const float* ppw    = (const float*)d_in[14];
    const float* ppb    = (const float*)d_in[15];
    const float* hw     = (const float*)d_in[16];
    const float* hbias  = (const float*)d_in[17];
    const float* ow     = (const float*)d_in[18];
    const float* ob     = (const float*)d_in[19];
    float* out = (float*)d_out;

    // workspace carve
    size_t off = 0;
    auto carve = [&](size_t bytes) {
        void* p = (char*)d_ws + off;
        off += (bytes + 255) & ~(size_t)255;
        return p;
    };
    ushort16* hb   = (ushort16*)carve((size_t)Nn * Hf * 2);
    ushort16* eb   = (ushort16*)carve((size_t)Ee * Hf * 2);
    float*    h    = (float*)carve((size_t)Nn * Hf * 4);
    float*    agg  = (float*)carve((size_t)Nn * Hf * 4);   // reused as pp
    uint32*   wq   = (uint32*)carve((size_t)Lf * 2 * 6144 * 4);
    uint32*   wpe  = (uint32*)carve((size_t)2048 * 4);
    float*    stats = (float*)carve(4 * 64 * 4);           // sums|sumsq|mu|rsig
    float*    sums = stats, *sumsq = stats + 64, *mu = stats + 128, *rsig = stats + 192;
    float*    poolb = (float*)carve((size_t)(Gf * Hf + Gf) * 4);
    float*    cnt   = poolb + (size_t)Gf * Hf;

    const int NH = Nn * Hf;                   // 3.2M

    // phase 0: embeds + weight swizzles
    node_embed_k<<<NH / 256, 256, 0, stream>>>(x, pnw, pnb, h, hb);
    pewswz_k<<<2048 / 256, 256, 0, stream>>>(pew, wpe);
    edge_embed_wmma_k<<<Ee / 64, 128, 0, stream>>>(ea, wpe, peb, eb);
    wswz_k<<<(Lf * 2 * 6144 + 255) / 256, 256, 0, stream>>>(csw, ccw, wq);

    // phase 1: message-passing layers (WMMA)
    for (int l = 0; l < Lf; ++l) {
        fill_k<<<NH / 256, 256, 0, stream>>>(agg, 0.f, NH);
        edge_conv_k<<<Ee / (CWAVES * TPW * 16), 128, 0, stream>>>(
            ei, hb, eb, wq + (size_t)l * 2 * 6144, csb + l * Hf, ccb + l * Hf, agg);
        fill_k<<<1, 128, 0, stream>>>(sums, 0.f, 128);   // sums + sumsq
        colstats_k<<<STAT_BLOCKS, 256, 0, stream>>>(agg, sums, sumsq);
        finstats_k<<<1, 64, 0, stream>>>(sums, sumsq, mu, rsig);
        bnres_k<<<NH / 256, 256, 0, stream>>>(agg, mu, rsig,
                                              gamma + l * Hf, beta + l * Hf, h, hb);
    }

    // phase 2: head
    prepool_k<<<NH / 256, 256, 0, stream>>>(h, ppw, ppb, agg);
    fill_k<<<(Gf * Hf + Gf + 255) / 256, 256, 0, stream>>>(poolb, 0.f, Gf * Hf + Gf);
    pool_k<<<NH / 256, 256, 0, stream>>>(agg, batch, poolb, cnt);
    head_k<<<Gf, 64, 0, stream>>>(poolb, cnt, hw, hbias, ow, ob, out);
}
